// MultiDecoder_54090818126396
// MI455X (gfx1250) — matrix operand order repaired and checked by reading, not compiled, not run on hardware
//
#include <hip/hip_runtime.h>
#include <hip/hip_bf16.h>

typedef __attribute__((ext_vector_type(16))) _Float16 v16h;
typedef __attribute__((ext_vector_type(8)))  _Float16 v8h;
typedef __attribute__((ext_vector_type(8)))  float    v8f;

#define NUM_A 10
#define CIN   12      // K of layer 1 (padded to 32)
#define H1N   128
#define H2N   256
#define FN    200     // N padded to 208 in packed weights
#define ROWS  32768   // B*T
#define LROW  120     // L

// ---- packed f16 weight layout in d_ws (units: halfs, per articulator) ----
// fragment = 32 lanes x 16 halfs = 512 halfs, stored [lane][e]
// B-fragment element mapping: N = nt*16 + (lane&15), K = kc*32 + (lane>>4)*16 + e
#define FRAG_H  512
#define L1_OFF  0        // 8 frags  (nt 0..7,  kc=0)
#define L2_OFF  4096     // 64 frags (nt*4+kc)
#define L3_OFF  36864    // 104 frags (nt*8+kc, nt 0..12)
#define PER_A   90112    // halfs per articulator

__global__ __launch_bounds__(256) void pack_weights(
    const float* __restrict__ W1, const float* __restrict__ W2,
    const float* __restrict__ W3, _Float16* __restrict__ wpk)
{
    int tid = blockIdx.x * 256 + threadIdx.x;
    if (tid >= NUM_A * PER_A) return;
    int a = tid / PER_A;
    int w = tid - a * PER_A;
    float v = 0.f;
    if (w < L2_OFF) {                       // layer 1: 12(->32) x 128
        int frag = w >> 9, within = w & 511;
        int lane = within >> 4, e = within & 15;
        int K = (lane >> 4) * 16 + e;
        int N = frag * 16 + (lane & 15);
        if (K < CIN) v = W1[a * CIN * H1N + K * H1N + N];
    } else if (w < L3_OFF) {                // layer 2: 128 x 256
        int w2 = w - L2_OFF;
        int frag = w2 >> 9, within = w2 & 511;
        int lane = within >> 4, e = within & 15;
        int nt = frag >> 2, kc = frag & 3;
        int K = kc * 32 + (lane >> 4) * 16 + e;
        int N = nt * 16 + (lane & 15);
        v = W2[a * H1N * H2N + K * H2N + N];
    } else {                                // layer 3: 256 x 200(->208)
        int w3 = w - L3_OFF;
        int frag = w3 >> 9, within = w3 & 511;
        int lane = within >> 4, e = within & 15;
        int nt = frag >> 3, kc = frag & 7;
        int K = kc * 32 + (lane >> 4) * 16 + e;
        int N = nt * 16 + (lane & 15);
        if (N < FN) v = W3[a * H2N * FN + K * FN + N];
    }
    wpk[tid] = (_Float16)v;
}

// ---- LDS staging: three buffers, padded strides for conflict-free b128 A reads
#define IN_STR 40     // 32 + 8 pad  (80 B/row  -> +20 banks/row, exact tiling)
#define H1_STR 136    // 128 + 8 pad (272 B/row -> +4 banks/row)
#define H2_STR 264    // 256 + 8 pad (528 B/row -> +4 banks/row)
#define IN_OFF 0
#define H1_OFF (64 * IN_STR)                // 2560
#define H2_OFF (H1_OFF + 64 * H1_STR)       // 11264
#define SMEM_H (H2_OFF + 64 * H2_STR)       // 28160 halfs = 56320 B

// 16-bit A-matrix 16x32 layout: lane L holds row M=L%16, half g=L/16;
// K = 16g + e (contiguous, matches packed B fragments)
static __device__ inline v16h lds_a_frag(const _Float16* base, int stride_h,
                                         int m, int g, int kc) {
    const _Float16* p = base + m * stride_h + kc * 32 + g * 16;
    v8h lo = *(const v8h*)p;          // K = kc*32 + 16g + 0..7
    v8h hi = *(const v8h*)(p + 8);    // K = kc*32 + 16g + 8..15
    v16h r;
#pragma unroll
    for (int i = 0; i < 8; ++i) { r[i] = lo[i]; r[i + 8] = hi[i]; }
    return r;
}

static __device__ inline v8f wmma16(v16h a, v16h b, v8f c) {
    return __builtin_amdgcn_wmma_f32_16x16x32_f16(
        false, a, false, b, (short)0, c, false, false);
}

// Block = 128 threads (4 waves) owns a 64-row tile of one articulator.
// Waves split the N dimension: each wave loads its B fragments ONCE and sweeps
// all four 16-row M sub-tiles -> 4x less B-fragment traffic from L2/WGP$.
__global__ __launch_bounds__(128) void decoder_fused(
    const float* __restrict__ x,
    const float* __restrict__ b1, const float* __restrict__ b2,
    const float* __restrict__ b3,
    const _Float16* __restrict__ wpk,
    float* __restrict__ out)
{
    __shared__ __align__(16) _Float16 sm[SMEM_H];
    const int tid = threadIdx.x;
    const int lane = tid & 31;
    const int wave = tid >> 5;
    const int a = blockIdx.y;
    const int rowBase = blockIdx.x * 64;
    const int g = lane >> 4;
    const int nloc = lane & 15;
    const _Float16* wpa = wpk + (size_t)a * PER_A;

    // ---- stage input tile as f16: 64 rows x 32 cols (cols 12..31 zero) ----
    for (int i = tid; i < 64 * 32; i += 128)
        sm[IN_OFF + (i >> 5) * IN_STR + (i & 31)] = (_Float16)0.f;
    for (int i = tid; i < 64 * CIN; i += 128) {
        int m = i / CIN, c = i - m * CIN;
        sm[IN_OFF + m * IN_STR + c] =
            (_Float16)x[(size_t)(rowBase + m) * LROW + a * CIN + c];
    }
    __syncthreads();

    // ---- layer 1: 64x12 * 12x128 ; wave handles nt = wave, wave+4 ----
    v16h ain[4];
#pragma unroll
    for (int mt = 0; mt < 4; ++mt)
        ain[mt] = lds_a_frag(sm + IN_OFF + mt * 16 * IN_STR, IN_STR, nloc, g, 0);
    for (int nt = wave; nt < 8; nt += 4) {
        float bias = b1[a * H1N + nt * 16 + nloc];
        v16h bf = *(const v16h*)(wpa + L1_OFF + nt * FRAG_H + lane * 16);
#pragma unroll
        for (int mt = 0; mt < 4; ++mt) {
            v8f acc;
#pragma unroll
            for (int r = 0; r < 8; ++r) acc[r] = bias;
            acc = wmma16(ain[mt], bf, acc);
#pragma unroll
            for (int r = 0; r < 8; ++r)
                sm[H1_OFF + (mt * 16 + r + 8 * g) * H1_STR + nt * 16 + nloc] =
                    (_Float16)fmaxf(acc[r], 0.f);
        }
    }
    __syncthreads();

    // ---- layer 2: 64x128 * 128x256 ; wave handles nt = wave,+4,+8,+12 ----
    for (int nt = wave; nt < 16; nt += 4) {
        float bias = b2[a * H2N + nt * 16 + nloc];
        const _Float16* bp = wpa + L2_OFF + (nt * 4) * FRAG_H + lane * 16;
        v16h bf[4];
#pragma unroll
        for (int kc = 0; kc < 4; ++kc)
            bf[kc] = *(const v16h*)(bp + kc * FRAG_H);
#pragma unroll
        for (int mt = 0; mt < 4; ++mt) {
            v8f acc;
#pragma unroll
            for (int r = 0; r < 8; ++r) acc[r] = bias;
            const _Float16* abase = sm + H1_OFF + mt * 16 * H1_STR;
#pragma unroll
            for (int kc = 0; kc < 4; ++kc)
                acc = wmma16(lds_a_frag(abase, H1_STR, nloc, g, kc), bf[kc], acc);
#pragma unroll
            for (int r = 0; r < 8; ++r)
                sm[H2_OFF + (mt * 16 + r + 8 * g) * H2_STR + nt * 16 + nloc] =
                    (_Float16)fmaxf(acc[r], 0.f);
        }
    }
    __syncthreads();

    // ---- layer 3: 64x256 * 256x200 ; wave handles nt = wave,+4,+8 (+12) ----
    for (int nt = wave; nt < 13; nt += 4) {
        int nglob = nt * 16 + nloc;
        float bias = (nglob < FN) ? b3[a * FN + nglob] : 0.f;
        const _Float16* bp = wpa + L3_OFF + (nt * 8) * FRAG_H + lane * 16;
        v16h bf[8];
#pragma unroll
        for (int kc = 0; kc < 8; ++kc)
            bf[kc] = *(const v16h*)(bp + kc * FRAG_H);
#pragma unroll
        for (int mt = 0; mt < 4; ++mt) {
            v8f acc;
#pragma unroll
            for (int r = 0; r < 8; ++r) acc[r] = bias;
            const _Float16* abase = sm + H2_OFF + mt * 16 * H2_STR;
#pragma unroll
            for (int kc = 0; kc < 8; ++kc)
                acc = wmma16(lds_a_frag(abase, H2_STR, nloc, g, kc), bf[kc], acc);
            if (nglob < FN) {
#pragma unroll
                for (int r = 0; r < 8; ++r)
                    out[(size_t)(rowBase + mt * 16 + r + 8 * g) * (NUM_A * FN) +
                        a * FN + nglob] = acc[r];
            }
        }
    }
}

extern "C" void kernel_launch(void* const* d_in, const int* in_sizes, int n_in,
                              void* d_out, int out_size, void* d_ws, size_t ws_size,
                              hipStream_t stream) {
    const float* x  = (const float*)d_in[0];
    const float* W1 = (const float*)d_in[1];
    const float* b1 = (const float*)d_in[2];
    const float* W2 = (const float*)d_in[3];
    const float* b2 = (const float*)d_in[4];
    const float* W3 = (const float*)d_in[5];
    const float* b3 = (const float*)d_in[6];
    // d_in[7] = idx, unused: it is the identity block mapping.
    _Float16* wpk = (_Float16*)d_ws;           // needs 10*90112*2 = 1,802,240 B
    float* out = (float*)d_out;

    int total = NUM_A * PER_A;
    pack_weights<<<(total + 255) / 256, 256, 0, stream>>>(W1, W2, W3, wpk);

    dim3 grid(ROWS / 64, NUM_A);               // 64 rows/block, 4 waves split N
    decoder_fused<<<grid, 128, 0, stream>>>(x, b1, b2, b3, wpk, out);
}
